// AttentionMask_58557584114274
// MI455X (gfx1250) — compile-verified, matrix operand order checked by reading, not run on hardware
//
#include <hip/hip_runtime.h>
#include <hip/hip_bf16.h>

// ---------------------------------------------------------------------------
// Masked multi-head attention, CDNA5 (gfx1250) WMMA pipeline, v2.
//   B=4, N=2048, C=768, H=12, D=64.
//   - All matmuls via v_wmma_f32_16x16x32_bf16 (fp32 accumulate).
//   - Double-buffered GLOBAL_LOAD_ASYNC_TO_LDS_B128 staging (ASYNCcnt) for
//     GEMM A/B tiles and for the attention K/V tiles (shared by all 8 waves
//     of a block -> 8x fewer global fetches).
//   - Flash-style streaming softmax: [B,H,N,N] scores never touch memory.
// ---------------------------------------------------------------------------

#define Bq   4
#define Nq   2048
#define Cq   768
#define Hq   12
#define Dq   64
#define BN   (Bq * Nq)          // 8192 rows
#define C3   (3 * Cq)           // 2304
#define NEGV (-65504.0f)

typedef __attribute__((ext_vector_type(16))) __bf16 bf16x16;
typedef __attribute__((ext_vector_type(8)))  __bf16 bf16x8;
typedef __attribute__((ext_vector_type(8)))  float  f32x8;

__device__ __forceinline__ f32x8 wmma_bf16(bf16x16 a, bf16x16 b, f32x8 c) {
  return __builtin_amdgcn_wmma_f32_16x16x32_bf16(
      /*neg_a=*/false, a, /*neg_b=*/false, b,
      /*c_mod=*/(short)0, c, /*reuse_a=*/false, /*reuse_b=*/false);
}

// --- CDNA5 async copy: one 16-byte chunk global -> LDS, tracked by ASYNCcnt.
__device__ __forceinline__ void async_copy16(void* lds_dst, const void* gsrc) {
  unsigned lo = (unsigned)(size_t)lds_dst;       // low 32 bits == LDS offset
  unsigned long long ga = (unsigned long long)(size_t)gsrc;
  asm volatile("global_load_async_to_lds_b128 %0, %1, off"
               :: "v"(lo), "v"(ga)
               : "memory");
}

__device__ __forceinline__ void async_wait0() {
#if __has_builtin(__builtin_amdgcn_s_wait_asynccnt)
  __builtin_amdgcn_s_wait_asynccnt(0);
#else
  asm volatile("s_wait_asynccnt 0x0" ::: "memory");
#endif
}

// 16x32 bf16 fragment from a row-major source (A layout; also the B layout
// when `row0` walks the 16 output columns of a W[Nout][K] matrix).
// ISA 16-bit layout: lane L holds row L%16, K elems {b..b+7, b+16..b+23},
// b = 8*(L>>4).  Works identically on global or LDS pointers.
__device__ __forceinline__ bf16x16 load_frag(const __bf16* __restrict__ base,
                                             int row0, int stride, int kcol,
                                             int lane) {
  const int lr = lane & 15;
  const int hb = (lane >> 4) * 8;
  const __bf16* p = base + (size_t)(row0 + lr) * stride + kcol + hb;
  bf16x8 lo = *(const bf16x8*)(p);
  bf16x8 hi = *(const bf16x8*)(p + 16);
  return __builtin_shufflevector(lo, hi, 0, 1, 2, 3, 4, 5, 6, 7,
                                 8, 9, 10, 11, 12, 13, 14, 15);
}

__device__ __forceinline__ float red_max16(float v) {
#pragma unroll
  for (int off = 8; off >= 1; off >>= 1) v = fmaxf(v, __shfl_xor(v, off, 32));
  return v;
}
__device__ __forceinline__ float red_sum16(float v) {
#pragma unroll
  for (int off = 8; off >= 1; off >>= 1) v += __shfl_xor(v, off, 32);
  return v;
}

// ---------------------------------------------------------------------------
// 1) conversions
// ---------------------------------------------------------------------------
__global__ void k_convert_xm(const float* __restrict__ x,
                             const float* __restrict__ mask,
                             __bf16* __restrict__ xm, int total) {
  int i = blockIdx.x * blockDim.x + threadIdx.x;
  if (i < total) xm[i] = (__bf16)(x[i] * mask[i / Cq]);
}

__global__ void k_convert(const float* __restrict__ w,
                          __bf16* __restrict__ o, int total) {
  int i = blockIdx.x * blockDim.x + threadIdx.x;
  if (i < total) o[i] = (__bf16)w[i];
}

// ---------------------------------------------------------------------------
// Shared GEMM core: block tile 128(M) x 128(N), K-step 32, 8 waves of 32x64.
// A[128][32] and B[128][32] bf16 tiles double-buffered in LDS via async copy.
// ---------------------------------------------------------------------------
#define KSTEPS (Cq / 32)   // 24

// stage one 128x32 bf16 tile (rows row0.., row stride `stride` elems)
__device__ __forceinline__ void stage_tile(__bf16* lds,               // [128][32]
                                           const __bf16* __restrict__ g,
                                           int row0, int stride, int k0,
                                           int tid) {
#pragma unroll
  for (int s = 0; s < 2; ++s) {
    const int c = tid + 256 * s;          // 512 x 16B chunks
    const int row = c >> 2;
    const int col = (c & 3) * 8;
    async_copy16(lds + row * 32 + col,
                 g + (size_t)(row0 + row) * stride + k0 + col);
  }
}

__device__ __forceinline__ bf16x16 lds_frag(const __bf16* tile, int row0,
                                            int lane) {
  const int lr = lane & 15;
  const int hb = (lane >> 4) * 8;
  const __bf16* p = tile + (row0 + lr) * 32 + hb;
  bf16x8 lo = *(const bf16x8*)(p);
  bf16x8 hi = *(const bf16x8*)(p + 16);
  return __builtin_shufflevector(lo, hi, 0, 1, 2, 3, 4, 5, 6, 7,
                                 8, 9, 10, 11, 12, 13, 14, 15);
}

// Computes the 2x4 f32x8 accumulators for this wave's 32x64 tile.
__device__ __forceinline__ void gemm_core(const __bf16* __restrict__ A,
                                          const __bf16* __restrict__ W,
                                          int m_blk, int n_blk, int tid,
                                          __bf16* sa /*[2][128][32]*/,
                                          __bf16* sb /*[2][128][32]*/,
                                          f32x8 acc[2][4]) {
  const int lane = tid & 31;
  const int wave = tid >> 5;
  const int wm = (wave & 3) * 32;        // wave row offset in block tile
  const int wn = (wave >> 2) * 64;       // wave col offset in block tile

  stage_tile(sa, A, m_blk, Cq, 0, tid);
  stage_tile(sb, W, n_blk, Cq, 0, tid);

  for (int kt = 0; kt < KSTEPS; ++kt) {
    async_wait0();
    __syncthreads();
    if (kt + 1 < KSTEPS) {
      __bf16* na = sa + ((kt + 1) & 1) * (128 * 32);
      __bf16* nb = sb + ((kt + 1) & 1) * (128 * 32);
      stage_tile(na, A, m_blk, Cq, (kt + 1) * 32, tid);
      stage_tile(nb, W, n_blk, Cq, (kt + 1) * 32, tid);
    }
    const __bf16* ta = sa + (kt & 1) * (128 * 32);
    const __bf16* tb = sb + (kt & 1) * (128 * 32);

    bf16x16 a0 = lds_frag(ta, wm, lane);
    bf16x16 a1 = lds_frag(ta, wm + 16, lane);
    bf16x16 b[4];
#pragma unroll
    for (int j = 0; j < 4; ++j) b[j] = lds_frag(tb, wn + 16 * j, lane);
#pragma unroll
    for (int j = 0; j < 4; ++j) {
      acc[0][j] = wmma_bf16(a0, b[j], acc[0][j]);
      acc[1][j] = wmma_bf16(a1, b[j], acc[1][j]);
    }
  }
}

// ---------------------------------------------------------------------------
// 2) QKV GEMM -> Q[B,H,N,D], K[B,H,N,D], Vt[B,H,D,N]
// ---------------------------------------------------------------------------
__global__ __launch_bounds__(256) void k_gemm_qkv(
    const __bf16* __restrict__ A, const __bf16* __restrict__ W,
    __bf16* __restrict__ Qh, __bf16* __restrict__ Kh,
    __bf16* __restrict__ Vt) {
  __shared__ __bf16 sa[2][128][32];
  __shared__ __bf16 sb[2][128][32];
  const int tid = threadIdx.x;
  const int lane = tid & 31;
  const int m_blk = blockIdx.y * 128;
  const int n_blk = blockIdx.x * 128;

  f32x8 acc[2][4] = {};
  gemm_core(A, W, m_blk, n_blk, tid, &sa[0][0][0], &sb[0][0][0], acc);

  const int wave = tid >> 5;
  const int m0 = m_blk + (wave & 3) * 32;
  const int n0 = n_blk + (wave >> 2) * 64;
  const int lr = lane & 15, hl = lane >> 4;
#pragma unroll
  for (int i = 0; i < 2; ++i)
#pragma unroll
    for (int j = 0; j < 4; ++j)
#pragma unroll
      for (int r = 0; r < 8; ++r) {
        const int gm = m0 + 16 * i + 8 * hl + r;   // row in [0,8192)
        const int gk = n0 + 16 * j + lr;           // col in [0,2304)
        const int b_ = gm >> 11, n_ = gm & (Nq - 1);
        const __bf16 v = (__bf16)acc[i][j][r];
        if (gk < Cq) {
          const int h_ = gk / Dq, d_ = gk % Dq;
          Qh[(((size_t)b_ * Hq + h_) * Nq + n_) * Dq + d_] = v;
        } else if (gk < 2 * Cq) {
          const int c_ = gk - Cq, h_ = c_ / Dq, d_ = c_ % Dq;
          Kh[(((size_t)b_ * Hq + h_) * Nq + n_) * Dq + d_] = v;
        } else {
          const int c_ = gk - 2 * Cq, h_ = c_ / Dq, d_ = c_ % Dq;
          Vt[(((size_t)b_ * Hq + h_) * Dq + d_) * Nq + n_] = v;
        }
      }
}

// ---------------------------------------------------------------------------
// 3) Flash attention.  8 waves x 16 query rows; 32 keys streamed per iter.
//    K tile [32 keys][64 d] and V tile [64 d][32 keys] async-staged in LDS,
//    shared by all 8 waves.  P re-laid-out C-frag -> A-frag via LDS.
// ---------------------------------------------------------------------------
__global__ __launch_bounds__(256) void k_attn(
    const __bf16* __restrict__ Qh, const __bf16* __restrict__ Kh,
    const __bf16* __restrict__ Vt, const float* __restrict__ mask,
    __bf16* __restrict__ AO) {
  __shared__ __bf16 kt_[2][32][64];   // 8 KB
  __shared__ __bf16 vt_[2][64][32];   // 8 KB
  __shared__ __bf16 pl[8][16][32];    // 8 KB per-wave P tiles

  const int tid = threadIdx.x;
  const int lane = tid & 31;
  const int wave = tid >> 5;
  const int lr = lane & 15, hl = lane >> 4;
  const int bh = blockIdx.y;
  const int b_ = bh / Hq, h_ = bh % Hq;
  const int q0 = blockIdx.x * 128 + wave * 16;

  const __bf16* Qb = Qh + (size_t)bh * Nq * Dq;
  const __bf16* Kb = Kh + (size_t)bh * Nq * Dq;
  const __bf16* Vb = Vt + (size_t)bh * Dq * Nq;
  const float* mb = mask + (size_t)b_ * Nq;

  bf16x16 qa0 = load_frag(Qb, q0, Dq, 0, lane);
  bf16x16 qa1 = load_frag(Qb, q0, Dq, 32, lane);

  float mq[8], rmax[8], rsum[8];
#pragma unroll
  for (int r = 0; r < 8; ++r) {
    mq[r] = mb[q0 + 8 * hl + r];
    rmax[r] = -1e30f;
    rsum[r] = 0.0f;
  }
  f32x8 o[4] = {};

  // cooperative K/V tile staging: 256 threads x 16B per tile
  const int krow = tid >> 3, kcol = (tid & 7) * 8;   // K: 32 rows x 128 B
  const int vrow = tid >> 2, vcol = (tid & 3) * 8;   // V: 64 rows x 64 B

  // prime buffer 0 (j0 = 0)
  async_copy16(&kt_[0][krow][kcol], Kb + (size_t)krow * Dq + kcol);
  async_copy16(&vt_[0][vrow][vcol], Vb + (size_t)vrow * Nq + vcol);

  const int iters = Nq / 32;  // 64
  for (int it = 0; it < iters; ++it) {
    const int j0 = it * 32;
    async_wait0();
    __syncthreads();
    if (it + 1 < iters) {
      const int jn0 = j0 + 32;
      const int nb = (it + 1) & 1;
      async_copy16(&kt_[nb][krow][kcol], Kb + (size_t)(jn0 + krow) * Dq + kcol);
      async_copy16(&vt_[nb][vrow][vcol], Vb + (size_t)vrow * Nq + jn0 + vcol);
    }
    const __bf16* ktile = &kt_[it & 1][0][0];
    const __bf16* vtile = &vt_[it & 1][0][0];

    // S = Q @ K^T : two 16x16 tiles over 32 keys, K-dim = D = 64
    f32x8 s[2] = {};
#pragma unroll
    for (int jt = 0; jt < 2; ++jt) {
      bf16x16 kb0 = load_frag(ktile, 16 * jt, Dq, 0, lane);
      bf16x16 kb1 = load_frag(ktile, 16 * jt, Dq, 32, lane);
      s[jt] = wmma_bf16(qa0, kb0, s[jt]);
      s[jt] = wmma_bf16(qa1, kb1, s[jt]);
    }

    const float mk0 = mb[j0 + lr];
    const float mk1 = mb[j0 + 16 + lr];
#pragma unroll
    for (int r = 0; r < 8; ++r) {
      float v0 = s[0][r] * 0.125f;  // D^-0.5
      float v1 = s[1][r] * 0.125f;
      if (mk0 == 0.0f || mq[r] == 0.0f) v0 = NEGV;
      if (mk1 == 0.0f || mq[r] == 0.0f) v1 = NEGV;
      s[0][r] = v0;
      s[1][r] = v1;
    }

    // online softmax (rows live across the 16-lane half-wave)
    float f[8];
#pragma unroll
    for (int r = 0; r < 8; ++r) {
      float t = red_max16(fmaxf(s[0][r], s[1][r]));
      float nm = fmaxf(rmax[r], t);
      f[r] = expf(rmax[r] - nm);
      rmax[r] = nm;
      float p0 = expf(s[0][r] - nm);
      float p1 = expf(s[1][r] - nm);
      s[0][r] = p0;
      s[1][r] = p1;
      rsum[r] = rsum[r] * f[r] + red_sum16(p0 + p1);
    }
#pragma unroll
    for (int jn = 0; jn < 4; ++jn)
#pragma unroll
      for (int r = 0; r < 8; ++r) o[jn][r] *= f[r];

    // C-frag -> A-frag re-layout of P via per-wave LDS region
    __syncthreads();
#pragma unroll
    for (int jt = 0; jt < 2; ++jt)
#pragma unroll
      for (int r = 0; r < 8; ++r)
        pl[wave][8 * hl + r][16 * jt + lr] = (__bf16)s[jt][r];
    __syncthreads();

    const __bf16* prow = &pl[wave][lr][0];
    bf16x8 plo = *(const bf16x8*)(prow + 8 * hl);
    bf16x8 phi = *(const bf16x8*)(prow + 16 + 8 * hl);
    bf16x16 pa = __builtin_shufflevector(plo, phi, 0, 1, 2, 3, 4, 5, 6, 7,
                                         8, 9, 10, 11, 12, 13, 14, 15);

    // O += P @ V from the staged V tile ([d][key_local], stride 32)
#pragma unroll
    for (int jn = 0; jn < 4; ++jn) {
      bf16x16 vb = lds_frag(vtile, 16 * jn, lane);
      o[jn] = wmma_bf16(pa, vb, o[jn]);
    }
  }

  // epilogue: normalize, post-softmax query mask, write [B,N,C] bf16
#pragma unroll
  for (int jn = 0; jn < 4; ++jn)
#pragma unroll
    for (int r = 0; r < 8; ++r) {
      const int m_ = q0 + 8 * hl + r;
      const int d_ = 16 * jn + lr;
      const float val = o[jn][r] / rsum[r] * mq[r];
      AO[((size_t)(b_ * Nq + m_)) * Cq + h_ * Dq + d_] = (__bf16)val;
    }
}

// ---------------------------------------------------------------------------
// 4) Proj GEMM: AO[8192,768]bf16 @ Wp[768,768]^T -> out fp32
// ---------------------------------------------------------------------------
__global__ __launch_bounds__(256) void k_gemm_proj(
    const __bf16* __restrict__ A, const __bf16* __restrict__ W,
    float* __restrict__ O) {
  __shared__ __bf16 sa[2][128][32];
  __shared__ __bf16 sb[2][128][32];
  const int tid = threadIdx.x;
  const int lane = tid & 31;
  const int m_blk = blockIdx.y * 128;
  const int n_blk = blockIdx.x * 128;

  f32x8 acc[2][4] = {};
  gemm_core(A, W, m_blk, n_blk, tid, &sa[0][0][0], &sb[0][0][0], acc);

  const int wave = tid >> 5;
  const int m0 = m_blk + (wave & 3) * 32;
  const int n0 = n_blk + (wave >> 2) * 64;
  const int lr = lane & 15, hl = lane >> 4;
#pragma unroll
  for (int i = 0; i < 2; ++i)
#pragma unroll
    for (int j = 0; j < 4; ++j)
#pragma unroll
      for (int r = 0; r < 8; ++r) {
        const int gm = m0 + 16 * i + 8 * hl + r;
        const int gk = n0 + 16 * j + lr;
        O[(size_t)gm * Cq + gk] = acc[i][j][r];
      }
}

// ---------------------------------------------------------------------------
// launch
// ---------------------------------------------------------------------------
extern "C" void kernel_launch(void* const* d_in, const int* in_sizes, int n_in,
                              void* d_out, int out_size, void* d_ws,
                              size_t ws_size, hipStream_t stream) {
  const float* x = (const float*)d_in[0];      // [B,N,C]
  const float* mask = (const float*)d_in[1];   // [B,N,1]
  const float* wqkv = (const float*)d_in[2];   // [3C,C]
  const float* wproj = (const float*)d_in[3];  // [C,C]
  float* out = (float*)d_out;                  // [B,N,C]

  char* ws = (char*)d_ws;
  size_t off = 0;
  auto carve = [&](size_t bytes) -> void* {
    void* p = ws + off;
    off += (bytes + 255) & ~(size_t)255;
    return p;
  };
  __bf16* XM = (__bf16*)carve((size_t)BN * Cq * 2);   // masked x, bf16
  __bf16* WQ = (__bf16*)carve((size_t)C3 * Cq * 2);   // w_qkv bf16
  __bf16* WP = (__bf16*)carve((size_t)Cq * Cq * 2);   // w_proj bf16
  __bf16* Qh = (__bf16*)carve((size_t)BN * Cq * 2);   // [B,H,N,D]
  __bf16* Kh = (__bf16*)carve((size_t)BN * Cq * 2);   // [B,H,N,D]
  __bf16* Vt = (__bf16*)carve((size_t)BN * Cq * 2);   // [B,H,D,N]
  __bf16* AO = (__bf16*)carve((size_t)BN * Cq * 2);   // attn out [B,N,C]

  const int tot_x = BN * Cq;
  k_convert_xm<<<(tot_x + 255) / 256, 256, 0, stream>>>(x, mask, XM, tot_x);
  k_convert<<<(C3 * Cq + 255) / 256, 256, 0, stream>>>(wqkv, WQ, C3 * Cq);
  k_convert<<<(Cq * Cq + 255) / 256, 256, 0, stream>>>(wproj, WP, Cq * Cq);

  k_gemm_qkv<<<dim3(C3 / 128, BN / 128), 256, 0, stream>>>(XM, WQ, Qh, Kh, Vt);
  k_attn<<<dim3(Nq / 128, Bq * Hq), 256, 0, stream>>>(Qh, Kh, Vt, mask, AO);
  k_gemm_proj<<<dim3(Cq / 128, BN / 128), 256, 0, stream>>>(AO, WP, out);
}
